// xLSTMEncoder_35519379538035
// MI455X (gfx1250) — matrix operand order repaired and checked by reading, not compiled, not run on hardware
//
#include <hip/hip_runtime.h>
#include <hip/hip_bf16.h>

typedef __attribute__((ext_vector_type(16))) _Float16 v16h;
typedef __attribute__((ext_vector_type(8)))  _Float16 v8h;
typedef __attribute__((ext_vector_type(8)))  float    v8f;
typedef __attribute__((ext_vector_type(4)))  float    v4f;

#define B_   128
#define S_   1500
#define SP_  1504          // padded to 94 full 16-row tiles
#define D_   128
#define NH_  4
#define HD_  32
#define FF_  192
#define EPS_ 1e-5f
#define NEG_ -1e30f
#define STILE 16
#define NT_   94

// ---------------------------------------------------------------- zero output
__global__ void k_zero(float* __restrict__ out) {
  int i = blockIdx.x * blockDim.x + threadIdx.x;
  if (i < B_ * D_) out[i] = 0.f;
}

// --------------------- weight prep: f32 -> f16 in WMMA B-fragment layout
// A fragment lane L: halves j -> K=(L>>4)*8+j (j<8), 16+(L>>4)*8+(j-8) (j>=8)
// B fragment lane L: half j -> K same as A fragment, N=(L&15)  (per 16x16 N-tile)
// frag storage: frag[tile*512 + lane*16 + j]  (one v16h per lane, coalesced)
__global__ void k_prep(const float* __restrict__ Wi, const float* __restrict__ Wf,
                       const float* __restrict__ Wz, const float* __restrict__ Wo,
                       const float* __restrict__ up, const float* __restrict__ dn,
                       _Float16* __restrict__ Wfrag, _Float16* __restrict__ upfrag,
                       _Float16* __restrict__ dnfrag) {
  const int NWT = 32 * 512;   // 4 gates * 4 heads * 2 ntiles
  const int NUP = 96 * 512;   // 4 kb * 24 nt
  const int NDN = 48 * 512;   // 6 kb * 8 nt
  int i = blockIdx.x * blockDim.x + threadIdx.x;
  if (i < NWT) {
    int tile = i >> 9, r = i & 511, lane = r >> 4, j = r & 15;
    int g = tile >> 3, head = (tile >> 1) & 3, nt = tile & 1;
    int K = (j < 8) ? (lane >> 4) * 8 + j : 16 + (lane >> 4) * 8 + (j - 8);   // d index
    int N = nt * 16 + (lane & 15);         // e index
    const float* W = (g == 0) ? Wi : (g == 1) ? Wf : (g == 2) ? Wz : Wo;
    Wfrag[i] = (_Float16)W[(head * HD_ + N) * HD_ + K];   // B[d][e] = W[h][e][d]
  } else if (i < NWT + NUP) {
    int q = i - NWT;
    int tile = q >> 9, r = q & 511, lane = r >> 4, j = r & 15;
    int kb = tile / 24, nt = tile % 24;
    int K = kb * 32 + ((j < 8) ? (lane >> 4) * 8 + j : 16 + (lane >> 4) * 8 + (j - 8));
    int N = nt * 16 + (lane & 15);
    upfrag[q] = (_Float16)up[K * (2 * FF_) + N];
  } else if (i < NWT + NUP + NDN) {
    int q = i - NWT - NUP;
    int tile = q >> 9, r = q & 511, lane = r >> 4, j = r & 15;
    int kb = tile / 8, nt = tile % 8;
    int K = kb * 32 + ((j < 8) ? (lane >> 4) * 8 + j : 16 + (lane >> 4) * 8 + (j - 8));
    int N = nt * 16 + (lane & 15);
    dnfrag[q] = (_Float16)dn[K * D_ + N];
  }
}

// ------------------------------------------- embedding + LN1 (one row/block)
__global__ void k_embed_ln(const int* __restrict__ seq, const float* __restrict__ emb,
                           const float* __restrict__ ln1w,
                           float* __restrict__ x, _Float16* __restrict__ h16) {
  __shared__ float r1[128], r2[128];
  int row = blockIdx.x, t = threadIdx.x;
  int b = row / S_, s = row % S_;
  int tok = seq[row];
  float v = emb[(size_t)tok * D_ + t];
  r1[t] = v; r2[t] = v * v;
  __syncthreads();
  for (int off = 64; off > 0; off >>= 1) {
    if (t < off) { r1[t] += r1[t + off]; r2[t] += r2[t + off]; }
    __syncthreads();
  }
  float mu  = r1[0] * (1.f / D_);
  float var = r2[0] * (1.f / D_) - mu * mu;
  float hv  = (v - mu) * rsqrtf(var + EPS_) * ln1w[t];
  size_t o = ((size_t)b * SP_ + s) * D_ + t;
  x[o] = v;
  h16[o] = (_Float16)hv;
}

// helper: build A fragment from a row-major [m][stride] f16 LDS tile
__device__ __forceinline__ v16h load_afrag(const _Float16* rowp, int k0, int base) {
  v8h lo = *(const v8h*)(rowp + k0 + base);
  v8h hi = *(const v8h*)(rowp + k0 + 16 + base);
  return __builtin_shufflevector(lo, hi, 0,1,2,3,4,5,6,7,8,9,10,11,12,13,14,15);
}

// -------------- causal depthwise conv + SiLU + WMMA gate projections
// gates layout: (B, SP_, 4, NH*HD) f32, gate order i,f,z,o (bias applied later)
__global__ void k_conv_gates(const _Float16* __restrict__ h16,
                             const float* __restrict__ convw, const float* __restrict__ convb,
                             const _Float16* __restrict__ Wfrag,
                             float* __restrict__ gates) {
  __shared__ __align__(16) _Float16 Axc[STILE][D_];
  __shared__ __align__(16) _Float16 Ah [STILE][D_];
  int tile = blockIdx.x, b = blockIdx.y, t = threadIdx.x;
  int s0 = tile * STILE;
  {   // channel c = t, rolling causal window of 4
    int c = t;
    float cw0 = convw[0 * D_ + c], cw1 = convw[1 * D_ + c];
    float cw2 = convw[2 * D_ + c], cw3 = convw[3 * D_ + c];
    float cb  = convb[c];
    auto ld = [&](int s) -> float {
      return (s >= 0 && s < S_) ? (float)h16[((size_t)b * SP_ + s) * D_ + c] : 0.f;
    };
    float w0 = ld(s0 - 3), w1 = ld(s0 - 2), w2 = ld(s0 - 1), w3 = ld(s0);
    for (int m = 0; m < STILE; ++m) {
      int s = s0 + m;
      float conv = w0 * cw0 + w1 * cw1 + w2 * cw2 + w3 * cw3 + cb;
      float sil  = conv / (1.f + __expf(-conv));
      bool ok = (s < S_);
      Axc[m][c] = ok ? (_Float16)sil : (_Float16)0.f;
      Ah [m][c] = ok ? (_Float16)w3  : (_Float16)0.f;
      w0 = w1; w1 = w2; w2 = w3; w3 = ld(s + 1);
    }
  }
  __syncthreads();
  // wave = gate (0:i from xc, 1:f from xc, 2:z from h, 3:o from h)
  int g = t >> 5, lane = t & 31;
  int mrow = lane & 15, hi = lane >> 4;
  int base = hi * 8, ncol = lane & 15;
  const _Float16 (*Asrc)[D_] = (g < 2) ? Axc : Ah;
  const v16h* Bf = (const v16h*)Wfrag;
  for (int head = 0; head < NH_; ++head) {
    v16h a = load_afrag(&Asrc[mrow][0], head * HD_, base);
    for (int nt = 0; nt < 2; ++nt) {
      v16h bm = Bf[((g * NH_ + head) * 2 + nt) * 32 + lane];
      v8f acc = {};
      acc = __builtin_amdgcn_wmma_f32_16x16x32_f16(false, a, false, bm,
                                                   (short)0, acc, false, false);
#pragma unroll
      for (int r = 0; r < 8; ++r) {
        int s = s0 + r + 8 * hi;    // < SP_ always (padded buffer)
        gates[(((size_t)b * SP_ + s) * 4 + g) * D_ + head * HD_ + nt * 16 + ncol] = acc[r];
      }
    }
  }
}

// ------------------- stabilized sLSTM scan + per-head group norm + residual
// one block per batch element; one wave per head; R packed [h][d][e][gate] in LDS
__global__ void k_recur(const float* __restrict__ gates,
                        const float* __restrict__ Ri, const float* __restrict__ Rf,
                        const float* __restrict__ Rz, const float* __restrict__ Ro,
                        const float* __restrict__ bi, const float* __restrict__ bf,
                        const float* __restrict__ bz, const float* __restrict__ bo,
                        const float* __restrict__ gnw,
                        float* __restrict__ x) {
  __shared__ __align__(16) float Rpk[NH_ * HD_ * HD_ * 4];   // 64 KB, gate-interleaved
  int b = blockIdx.x, t = threadIdx.x;
  for (int i = t; i < NH_ * HD_ * HD_; i += 128) {
    Rpk[i * 4 + 0] = Ri[i]; Rpk[i * 4 + 1] = Rf[i];
    Rpk[i * 4 + 2] = Rz[i]; Rpk[i * 4 + 3] = Ro[i];
  }
  __syncthreads();
  int h = t >> 5, e = t & 31;
  float bie = bi[h * HD_ + e], bfe = bf[h * HD_ + e];
  float bze = bz[h * HD_ + e], boe = bo[h * HD_ + e];
  float gwe = gnw[h * HD_ + e];
  const float* Rh = &Rpk[h * HD_ * HD_ * 4];
  float y = 0.f, c = 0.f, n = 0.f, m = NEG_;
  for (int s = 0; s < S_; ++s) {
    size_t gb = ((size_t)b * SP_ + s) * 4 * D_ + h * HD_ + e;
    float gi = gates[gb], gf = gates[gb + D_];
    float gz = gates[gb + 2 * D_], go = gates[gb + 3 * D_];
    if (s + 1 < S_) __builtin_prefetch(&gates[gb + 4 * D_], 0, 1);
    float ai = 0.f, af = 0.f, az = 0.f, ao = 0.f;
#pragma unroll
    for (int d = 0; d < HD_; ++d) {
      float yd = __shfl(y, d, 32);
      v4f rc = *(const v4f*)&Rh[(d * HD_ + e) * 4];   // one ds_load_b128
      ai = fmaf(yd, rc[0], ai);
      af = fmaf(yd, rc[1], af);
      az = fmaf(yd, rc[2], az);
      ao = fmaf(yd, rc[3], ao);
    }
    float ir  = gi + ai + bie;
    float fr  = gf + af + bfe;
    float zr  = gz + az + bze;
    float orr = go + ao + boe;
    float ls = (fr >= 0.f) ? -log1pf(__expf(-fr)) : (fr - log1pf(__expf(fr)));
    float lf = m + ls;
    float mn = fmaxf(ir, lf);
    float ig = __expf(ir - mn);
    float fg = __expf(lf - mn);
    float cn = fg * c + ig * tanhf(zr);
    float nn = fg * n + ig;
    float yo = (1.f / (1.f + __expf(-orr))) * cn / nn;
    y = yo; c = cn; n = nn; m = mn;
    // group norm over the 32 lanes of this head
    float sv = yo;
    for (int o = 16; o > 0; o >>= 1) sv += __shfl_xor(sv, o, 32);
    float mu = sv * (1.f / HD_);
    float dv = yo - mu;
    float sq = dv * dv;
    for (int o = 16; o > 0; o >>= 1) sq += __shfl_xor(sq, o, 32);
    float yg = dv * rsqrtf(sq * (1.f / HD_) + EPS_) * gwe;
    size_t xi = ((size_t)b * SP_ + s) * D_ + h * HD_ + e;
    x[xi] += yg;                  // residual, in place
  }
}

// ---------------- LN2 + GEGLU FFN (WMMA) + residual + post-LN + mean pool
__global__ void k_ffn_pool(const float* __restrict__ x, const float* __restrict__ ln2w,
                           const _Float16* __restrict__ upfrag,
                           const _Float16* __restrict__ dnfrag,
                           const float* __restrict__ lnpw,
                           float* __restrict__ out) {
  __shared__ float    xrow[STILE][D_];
  __shared__ __align__(16) _Float16 h2t[STILE][D_];
  __shared__ float    ut  [STILE][2 * FF_];
  __shared__ __align__(16) _Float16 act[STILE][FF_];
  __shared__ float    x3t [STILE][D_];
  __shared__ float    red1[STILE][8], red2[STILE][8];
  __shared__ float    muA[STILE], rsA[STILE];
  int tile = blockIdx.x, b = blockIdx.y, t = threadIdx.x;
  int s0 = tile * STILE;
  for (int m = 0; m < STILE; ++m)    // padded buffer: unconditional load
    xrow[m][t] = x[((size_t)b * SP_ + s0 + m) * D_ + t];
  __syncthreads();
  {   // LN2 row stats: 8 threads per row
    int r = t >> 3, g8 = t & 7;
    float s1 = 0.f, s2 = 0.f;
    for (int j = 0; j < 16; ++j) { float v = xrow[r][g8 * 16 + j]; s1 += v; s2 += v * v; }
    red1[r][g8] = s1; red2[r][g8] = s2;
  }
  __syncthreads();
  if (t < STILE) {
    float s1 = 0.f, s2 = 0.f;
    for (int j = 0; j < 8; ++j) { s1 += red1[t][j]; s2 += red2[t][j]; }
    float mu = s1 * (1.f / D_);
    muA[t] = mu; rsA[t] = rsqrtf(s2 * (1.f / D_) - mu * mu + EPS_);
  }
  __syncthreads();
  for (int m = 0; m < STILE; ++m)
    h2t[m][t] = (_Float16)((xrow[m][t] - muA[m]) * rsA[m] * ln2w[t]);
  __syncthreads();
  int w = t >> 5, lane = t & 31;
  int mrow = lane & 15, hi = lane >> 4;
  int base = hi * 8, ncol = lane & 15;
  const v16h* Uf = (const v16h*)upfrag;
  const v16h* Df = (const v16h*)dnfrag;
  // GEMM1: (16 x 128) @ (128 x 384)
  for (int nt = w; nt < 2 * FF_ / 16; nt += 4) {
    v8f acc = {};
    for (int kb = 0; kb < D_ / 32; ++kb) {
      v16h a  = load_afrag(&h2t[mrow][0], kb * 32, base);
      v16h bm = Uf[(kb * 24 + nt) * 32 + lane];
      acc = __builtin_amdgcn_wmma_f32_16x16x32_f16(false, a, false, bm,
                                                   (short)0, acc, false, false);
    }
#pragma unroll
    for (int r = 0; r < 8; ++r) ut[r + 8 * hi][nt * 16 + ncol] = acc[r];
  }
  __syncthreads();
  // GEGLU: gelu(u[:,:FF]) * u[:,FF:]
  for (int idx = t; idx < STILE * FF_; idx += 128) {
    int m = idx / FF_, f = idx % FF_;
    float ug = ut[m][f], ul = ut[m][FF_ + f];
    float gel = 0.5f * ug * (1.f + erff(ug * 0.70710678118654752f));
    act[m][f] = (_Float16)(gel * ul);
  }
  __syncthreads();
  // GEMM2: (16 x 192) @ (192 x 128) + residual
  for (int nt = w; nt < D_ / 16; nt += 4) {
    v8f acc = {};
    for (int kb = 0; kb < FF_ / 32; ++kb) {
      v16h a  = load_afrag(&act[mrow][0], kb * 32, base);
      v16h bm = Df[(kb * 8 + nt) * 32 + lane];
      acc = __builtin_amdgcn_wmma_f32_16x16x32_f16(false, a, false, bm,
                                                   (short)0, acc, false, false);
    }
#pragma unroll
    for (int r = 0; r < 8; ++r) {
      int m = r + 8 * hi;
      x3t[m][nt * 16 + ncol] = acc[r] + xrow[m][nt * 16 + ncol];
    }
  }
  __syncthreads();
  {   // post-LN stats
    int r = t >> 3, g8 = t & 7;
    float s1 = 0.f, s2 = 0.f;
    for (int j = 0; j < 16; ++j) { float v = x3t[r][g8 * 16 + j]; s1 += v; s2 += v * v; }
    red1[r][g8] = s1; red2[r][g8] = s2;
  }
  __syncthreads();
  if (t < STILE) {
    float s1 = 0.f, s2 = 0.f;
    for (int j = 0; j < 8; ++j) { s1 += red1[t][j]; s2 += red2[t][j]; }
    float mu = s1 * (1.f / D_);
    muA[t] = mu; rsA[t] = rsqrtf(s2 * (1.f / D_) - mu * mu + EPS_);
  }
  __syncthreads();
  float sum = 0.f;
  for (int m = 0; m < STILE; ++m) {
    int s = s0 + m;
    if (s < S_) sum += (x3t[m][t] - muA[m]) * rsA[m];   // pad rows excluded
  }
  atomicAdd(&out[b * D_ + t], sum * lnpw[t] * (1.f / S_));
}

// ---------------------------------------------------------------------------
extern "C" void kernel_launch(void* const* d_in, const int* in_sizes, int n_in,
                              void* d_out, int out_size, void* d_ws, size_t ws_size,
                              hipStream_t stream) {
  const int*   seq   = (const int*)  d_in[0];
  const float* emb   = (const float*)d_in[1];
  const float* convw = (const float*)d_in[2];
  const float* convb = (const float*)d_in[3];
  const float* Wi    = (const float*)d_in[4];
  const float* Wf    = (const float*)d_in[5];
  const float* Wz    = (const float*)d_in[6];
  const float* Wo    = (const float*)d_in[7];
  const float* Ri    = (const float*)d_in[8];
  const float* Rf    = (const float*)d_in[9];
  const float* Rz    = (const float*)d_in[10];
  const float* Ro    = (const float*)d_in[11];
  const float* bi    = (const float*)d_in[12];
  const float* bf    = (const float*)d_in[13];
  const float* bz    = (const float*)d_in[14];
  const float* bo    = (const float*)d_in[15];
  const float* gnw   = (const float*)d_in[16];
  const float* ln1w  = (const float*)d_in[17];
  const float* ln2w  = (const float*)d_in[18];
  const float* upw   = (const float*)d_in[19];
  const float* dnw   = (const float*)d_in[20];
  const float* lnpw  = (const float*)d_in[21];
  (void)in_sizes; (void)n_in; (void)out_size; (void)ws_size;

  // workspace layout (~543 MB), S padded to 1504 rows
  char* ws = (char*)d_ws;
  size_t off = 0;
  float*    x     = (float*)(ws + off);    off += (size_t)B_ * SP_ * D_ * 4;     // 98.6 MB
  _Float16* h16   = (_Float16*)(ws + off); off += (size_t)B_ * SP_ * D_ * 2;     // 49.3 MB
  float*    gates = (float*)(ws + off);    off += (size_t)B_ * SP_ * 4 * D_ * 4; // 394 MB
  _Float16* Wfrag = (_Float16*)(ws + off); off += (size_t)32 * 512 * 2;
  _Float16* upfrg = (_Float16*)(ws + off); off += (size_t)96 * 512 * 2;
  _Float16* dnfrg = (_Float16*)(ws + off); off += (size_t)48 * 512 * 2;

  float* out = (float*)d_out;
  k_zero<<<(B_ * D_ + 255) / 256, 256, 0, stream>>>(out);
  k_prep<<<((32 + 96 + 48) * 512 + 255) / 256, 256, 0, stream>>>(
      Wi, Wf, Wz, Wo, upw, dnw, Wfrag, upfrg, dnfrg);
  k_embed_ln<<<B_ * S_, 128, 0, stream>>>(seq, emb, ln1w, x, h16);
  k_conv_gates<<<dim3(NT_, B_), 128, 0, stream>>>(h16, convw, convb, Wfrag, gates);
  k_recur<<<B_, 128, 0, stream>>>(gates, Ri, Rf, Rz, Ro, bi, bf, bz, bo, gnw, x);
  k_ffn_pool<<<dim3(NT_, B_), 128, 0, stream>>>(x, ln2w, upfrg, dnfrg, lnpw, out);
}
